// SeqAttention_72310069395510
// MI455X (gfx1250) — matrix operand order repaired and checked by reading, not compiled, not run on hardware
//
#include <hip/hip_runtime.h>

#define B_   4
#define SEQ  2048
#define SPAN 4096
#define DM   512

typedef __bf16 bf16t;
typedef __attribute__((ext_vector_type(16))) __bf16 v16bf;
typedef __attribute__((ext_vector_type(8)))  __bf16 v8bf;
typedef __attribute__((ext_vector_type(8)))  float  v8f;

__device__ __forceinline__ unsigned pk2(float a, float b) {
  unsigned short ua = __builtin_bit_cast(unsigned short, (bf16t)a);
  unsigned short ub = __builtin_bit_cast(unsigned short, (bf16t)b);
  return (unsigned)ua | ((unsigned)ub << 16);
}

// A-fragment (16-bit A layout): per-lane chunks at +0 and +16 elements from base
__device__ __forceinline__ v16bf load_afrag(const bf16t* p) {
  v8bf lo = *(const v8bf*)(p);
  v8bf hi = *(const v8bf*)(p + 16);
  return __builtin_shufflevector(lo, hi, 0,1,2,3,4,5,6,7,8,9,10,11,12,13,14,15);
}

__global__ void zero_loss_k(float* loss) {
  if (threadIdx.x < B_) loss[threadIdx.x] = 0.f;
}

// r = sigmoid(mem . w_span + b_span) * MEM_LEN - counter ; closed-form ramp loss
__global__ void span_k(const float* __restrict__ mem, const float* __restrict__ counter,
                       const float* __restrict__ w_span, const float* __restrict__ b_span,
                       float* __restrict__ r_out, float* __restrict__ loss_out) {
  const int lane = threadIdx.x & 31;
  const int gw = blockIdx.x * (blockDim.x >> 5) + (threadIdx.x >> 5);   // one wave per (b,j)
  const int b = gw >> 12;
  const float* mrow = mem + (size_t)gw * DM;
  float acc = 0.f;
#pragma unroll
  for (int d0 = 0; d0 < DM; d0 += 128) {
    const int d = d0 + lane * 4;
    float4 mv = *(const float4*)(mrow + d);
    float4 wv = *(const float4*)(w_span + d);
    acc += mv.x * wv.x + mv.y * wv.y + mv.z * wv.z + mv.w * wv.w;
  }
#pragma unroll
  for (int off = 16; off >= 1; off >>= 1) acc += __shfl_xor(acc, off);
  const float x = acc + b_span[0];
  const float e = 2048.f / (1.f + __expf(-x));
  const float r = e - counter[gw];
  if (lane == 0) {
    r_out[gw] = r;
    // sum of (r - i) over integer i in (r, r+128) intersect [0, 2048)
    int lo = (int)floorf(r) + 1; if (lo < 0) lo = 0;
    int hi = (int)ceilf(r + 128.f); if (hi > SEQ) hi = SEQ;
    if (hi > lo) {
      const float n = (float)(hi - lo);
      const float sumi = 0.5f * (float)(lo + hi - 1) * n;
      atomicAdd(loss_out + b, (n * r - sumi) * (1e-6f / 2048.f));
    }
  }
}

// Flash-style attention with expire-span weighting.
// Grid: (SEQ/64, B), 128 threads (4 waves). Wave w owns q-rows [i0+16w, i0+16w+16).
__global__ __launch_bounds__(128)
__attribute__((amdgpu_waves_per_eu(1, 2)))
void attn_k(const float* __restrict__ qg, const float* __restrict__ kg,
            const float* __restrict__ vg, const float* __restrict__ rbuf,
            float* __restrict__ out) {
  extern __shared__ char smem_raw[];
  bf16t* Qs = (bf16t*)smem_raw;     // [64][512]
  bf16t* Ks = Qs + 64 * DM;         // [32][512]
  bf16t* Vs = Ks + 32 * DM;         // transposed [512][32]
  bf16t* Ps = Vs + DM * 32;         // [4][16][32]

  const int tid  = threadIdx.x;
  const int lane = tid & 31;
  const int wid  = tid >> 5;
  const int b    = blockIdx.y;
  const int i0   = blockIdx.x * 64;
  const int hl   = lane >> 4;       // half-wave select
  const int ll   = lane & 15;

  { // stage Q tile (64 rows), f32 -> bf16, once
    const float* qb = qg + (((size_t)b * SEQ) + i0) * DM;
    unsigned* Qu = (unsigned*)Qs;
    for (int e = tid * 4; e < 64 * DM; e += 128 * 4) {
      float4 t = *(const float4*)(qb + e);
      Qu[(e >> 1) + 0] = pk2(t.x, t.y);
      Qu[(e >> 1) + 1] = pk2(t.z, t.w);
    }
  }

  v8f acc[32] = {};
  v8f Mv, Lv, Zv;
#pragma unroll
  for (int i = 0; i < 8; ++i) { Mv[i] = -1e30f; Lv[i] = 0.f; Zv[i] = 0.f; }

  const float SCALE = 0.044194173824159216f; // 1/sqrt(512)
  const int ibase = i0 + wid * 16 + (hl << 3);
  const int nsteps = (i0 + 2112) >> 5;       // causal: j <= i + 2048

  // A-frag base (16 contiguous at +16*hl); B-frag base (16 contiguous at +16*hl)
  const bf16t* qfb  = Qs + (wid * 16 + ll) * DM + (hl << 4);
  const bf16t* kfb0 = Ks + ll * DM + (hl << 4);
  const bf16t* kfb1 = Ks + (16 + ll) * DM + (hl << 4);
  const bf16t* vfb  = Vs + ll * 32 + (hl << 4);

  for (int s = 0; s < nsteps; ++s) {
    const int j0 = s << 5;
    { // stage K row-major + V transposed, f32 -> bf16
      const float* kb = kg + (((size_t)b * SPAN) + j0) * DM;
      unsigned* Ku = (unsigned*)Ks;
      for (int e = tid * 4; e < 32 * DM; e += 128 * 4) {
        float4 t = *(const float4*)(kb + e);
        Ku[(e >> 1) + 0] = pk2(t.x, t.y);
        Ku[(e >> 1) + 1] = pk2(t.z, t.w);
        __builtin_prefetch(kb + 32 * DM + e, 0, 1);
      }
      const float* vb = vg + (((size_t)b * SPAN) + j0) * DM;
      for (int e = tid * 4; e < 32 * DM; e += 128 * 4) {
        const int jj = e >> 9, col = e & 511;
        float4 t = *(const float4*)(vb + e);
        Vs[(col + 0) * 32 + jj] = (bf16t)t.x;
        Vs[(col + 1) * 32 + jj] = (bf16t)t.y;
        Vs[(col + 2) * 32 + jj] = (bf16t)t.z;
        Vs[(col + 3) * 32 + jj] = (bf16t)t.w;
        __builtin_prefetch(vb + 32 * DM + e, 0, 1);
      }
    }
    __syncthreads();

    // ---- phase 1: S(16x32) = Q(16x512) . K_tile^T ----
    // Grouped: both k-substeps' fragments live before the 4-WMMA burst.
    v8f s0 = {}, s1 = {};
#pragma unroll
    for (int kk = 0; kk < DM; kk += 64) {
      v16bf a0  = *(const v16bf*)(qfb + kk);
      v16bf b00 = *(const v16bf*)(kfb0 + kk);
      v16bf b10 = *(const v16bf*)(kfb1 + kk);
      v16bf a1  = *(const v16bf*)(qfb + kk + 32);
      v16bf b01 = *(const v16bf*)(kfb0 + kk + 32);
      v16bf b11 = *(const v16bf*)(kfb1 + kk + 32);
      s0 = __builtin_amdgcn_wmma_f32_16x16x32_bf16(false, a0, false, b00, (short)0, s0, false, false);
      s1 = __builtin_amdgcn_wmma_f32_16x16x32_bf16(false, a0, false, b10, (short)0, s1, false, false);
      s0 = __builtin_amdgcn_wmma_f32_16x16x32_bf16(false, a1, false, b01, (short)0, s0, false, false);
      s1 = __builtin_amdgcn_wmma_f32_16x16x32_bf16(false, a1, false, b11, (short)0, s1, false, false);
    }

    // ---- online softmax with expire-span weight m ----
    const float r0 = rbuf[(size_t)b * SPAN + j0 + ll];
    const float r1 = rbuf[(size_t)b * SPAN + j0 + 16 + ll];
    const int jg0 = j0 + ll, jg1 = j0 + 16 + ll;
    bf16t* pw = Ps + wid * 512;
    v8f scv;
    float needs = 0.f;
#pragma unroll
    for (int vv = 0; vv < 8; ++vv) {
      const int irow = ibase + vv;
      float m0 = fminf(fmaxf((r0 - (float)irow) * (1.f / 128.f) + 1.f, 0.f), 1.f);
      float m1 = fminf(fmaxf((r1 - (float)irow) * (1.f / 128.f) + 1.f, 0.f), 1.f);
      if (jg0 > irow + 2048) m0 = 0.f;
      if (jg1 > irow + 2048) m1 = 0.f;
      const float x0 = (m0 > 0.f) ? s0[vv] * SCALE : -3e38f;
      const float x1 = (m1 > 0.f) ? s1[vv] * SCALE : -3e38f;
      float mx = fmaxf(x0, x1);
      mx = fmaxf(mx, __shfl_xor(mx, 1));
      mx = fmaxf(mx, __shfl_xor(mx, 2));
      mx = fmaxf(mx, __shfl_xor(mx, 4));
      mx = fmaxf(mx, __shfl_xor(mx, 8));
      const float Mn = fmaxf(Mv[vv], mx);
      const float sc = __expf(Mv[vv] - Mn);
      const float z0 = __expf(x0 - Mn), z1 = __expf(x1 - Mn);
      const float w0 = m0 * z0, w1 = m1 * z1;
      float zs = z0 + z1, ws = w0 + w1;
      zs += __shfl_xor(zs, 1); zs += __shfl_xor(zs, 2);
      zs += __shfl_xor(zs, 4); zs += __shfl_xor(zs, 8);
      ws += __shfl_xor(ws, 1); ws += __shfl_xor(ws, 2);
      ws += __shfl_xor(ws, 4); ws += __shfl_xor(ws, 8);
      Lv[vv] = Lv[vv] * sc + ws;
      Zv[vv] = Zv[vv] * sc + zs;
      Mv[vv] = Mn;
      scv[vv] = sc;
      needs += (sc != 1.f) ? 1.f : 0.f;
      const int prow = vv + (hl << 3);          // C-layout -> LDS (A layout source)
      pw[prow * 32 + ll] = (bf16t)w0;
      pw[prow * 32 + 16 + ll] = (bf16t)w1;
    }
    if (__ballot(needs != 0.f)) {
#pragma unroll
      for (int n = 0; n < 32; ++n)
#pragma unroll
        for (int vv = 0; vv < 8; ++vv)
          acc[n][vv] *= scv[vv];
    }
    asm volatile("s_wait_dscnt 0" ::: "memory"); // wave-local P visibility

    // ---- phase 2: O(16x512) += P(16x32) . V(32x512) ----
    // Grouped: 4 B-fragments live before the 4-WMMA burst.
    {
      v16bf pa = *(const v16bf*)(pw + ll * 32 + (hl << 4));
#pragma unroll
      for (int n4 = 0; n4 < 32; n4 += 4) {
        v16bf f0 = *(const v16bf*)(vfb + (n4 + 0) * 512);
        v16bf f1 = *(const v16bf*)(vfb + (n4 + 1) * 512);
        v16bf f2 = *(const v16bf*)(vfb + (n4 + 2) * 512);
        v16bf f3 = *(const v16bf*)(vfb + (n4 + 3) * 512);
        acc[n4 + 0] = __builtin_amdgcn_wmma_f32_16x16x32_bf16(false, pa, false, f0, (short)0, acc[n4 + 0], false, false);
        acc[n4 + 1] = __builtin_amdgcn_wmma_f32_16x16x32_bf16(false, pa, false, f1, (short)0, acc[n4 + 1], false, false);
        acc[n4 + 2] = __builtin_amdgcn_wmma_f32_16x16x32_bf16(false, pa, false, f2, (short)0, acc[n4 + 2], false, false);
        acc[n4 + 3] = __builtin_amdgcn_wmma_f32_16x16x32_bf16(false, pa, false, f3, (short)0, acc[n4 + 3], false, false);
      }
    }
    __syncthreads();
  }

  // epilogue: out = acc / (L + 1e-8 * Z)
#pragma unroll
  for (int vv = 0; vv < 8; ++vv) {
    const float denom = Lv[vv] + 1e-8f * Zv[vv];
    const float inv = (denom > 0.f) ? (1.f / denom) : 0.f;
    const int irow = ibase + vv;
    float* orow = out + (((size_t)b * SEQ) + irow) * DM + ll;
#pragma unroll
    for (int n = 0; n < 32; ++n)
      orow[n * 16] = acc[n][vv] * inv;
  }
}

extern "C" void kernel_launch(void* const* d_in, const int* in_sizes, int n_in,
                              void* d_out, int out_size, void* d_ws, size_t ws_size,
                              hipStream_t stream) {
  const float* q       = (const float*)d_in[0];
  const float* k       = (const float*)d_in[1];
  const float* v       = (const float*)d_in[2];
  const float* mem     = (const float*)d_in[3];
  const float* counter = (const float*)d_in[4];
  const float* w_span  = (const float*)d_in[5];
  const float* b_span  = (const float*)d_in[6];

  float* out   = (float*)d_out;
  float* r_out = out + (size_t)B_ * SEQ * DM;     // second output
  float* loss  = r_out + (size_t)B_ * SPAN;       // third output

  zero_loss_k<<<1, 64, 0, stream>>>(loss);
  span_k<<<(B_ * SPAN) / 8, 256, 0, stream>>>(mem, counter, w_span, b_span, r_out, loss);

  const size_t smem = (size_t)(64 * DM + 32 * DM + DM * 32 + 4 * 16 * 32) * sizeof(bf16t); // 132 KB
  attn_k<<<dim3(SEQ / 64, B_), 128, smem, stream>>>(q, k, v, r_out, out);
}
